// MOTORTaskHead_60936995995890
// MI455X (gfx1250) — matrix operand order repaired and checked by reading, not compiled, not run on hardware
//
#include <hip/hip_runtime.h>
#include <hip/hip_bf16.h>
#include <math.h>

typedef _Float16 f16;
typedef __attribute__((ext_vector_type(16))) _Float16 v16h;
typedef __attribute__((ext_vector_type(8)))  _Float16 v8h;
typedef __attribute__((ext_vector_type(8)))  float    v8f;

#define B_DIM 512
#define T_DIM 8
#define K_DIM 8192
#define H_DIM 768
#define F_DIM 512
#define EPS_C      1e-8f
#define ONE_M_EPS  (1.0f - 1e-8f)
#define RMS_EPS_C  1e-6f
#define LN_T       2.0794415416798357f   // ln(8)

// ---------------------------------------------------------------------------
// WMMA fragment loaders (CDNA5 16-bit layouts, wave32).
// A (16x32, MxK): lane(0..15)=row M, lane+16 same row; per-lane halves:
//   h in [0,8)  -> k = half*8 + h ; h in [8,16) -> k = 16 + half*8 + (h-8)
// B (32x16, KxN): lane(0..15)=col N; per-lane k same as A (table order).
// ---------------------------------------------------------------------------
__device__ static inline v16h load_a_frag(const f16* __restrict__ row_k0, int half) {
  const v8h c0 = *(const v8h*)(row_k0 + half * 8);
  const v8h c1 = *(const v8h*)(row_k0 + 16 + half * 8);
  v16h a;
#pragma unroll
  for (int i = 0; i < 8; ++i) { a[i] = c0[i]; a[i + 8] = c1[i]; }
  return a;
}

__device__ static inline v16h load_b_frag(const f16* __restrict__ col_k0, int half) {
  const v8h c0 = *(const v8h*)(col_k0 + half * 8);
  const v8h c1 = *(const v8h*)(col_k0 + 16 + half * 8);
  v16h b;
#pragma unroll
  for (int i = 0; i < 8; ++i) { b[i] = c0[i]; b[i + 8] = c1[i]; }
  return b;
}

__device__ static inline v8f wmma_f16(v16h a, v16h b, v8f c) {
  return __builtin_amdgcn_wmma_f32_16x16x32_f16(false, a, false, b, (short)0, c,
                                                false, false);
}

// ---------------------------------------------------------------------------
// Converters: f32 -> f16 (plain and transposed).
// ---------------------------------------------------------------------------
__global__ void cvt_f16(const float* __restrict__ s, f16* __restrict__ d, int n) {
  for (int i = blockIdx.x * blockDim.x + threadIdx.x; i < n;
       i += gridDim.x * blockDim.x)
    d[i] = (f16)s[i];
}

// s: rows x cols (row-major) -> d: cols x rows (row-major). Reads coalesced.
__global__ void cvt_f16_t(const float* __restrict__ s, f16* __restrict__ d,
                          int rows, int cols) {
  const int n = rows * cols;
  for (int i = blockIdx.x * blockDim.x + threadIdx.x; i < n;
       i += gridDim.x * blockDim.x) {
    const int c = i % cols, r = i / cols;
    d[c * rows + r] = (f16)s[i];
  }
}

// ---------------------------------------------------------------------------
// Kernel A: x = features @ W1 + b1, fused RMS-norm * norm_w, emit f16 xn
// laid out as GEMM2's A matrix: xn[(b*T + t)*F + f].
// Grid: (B/16, T). 256 threads = 8 waves; wave w owns 4 N-tiles (64 cols).
// ---------------------------------------------------------------------------
__global__ __launch_bounds__(256) void gemm1_rms(
    const f16* __restrict__ feath,   // B x H row-major
    const f16* __restrict__ w1t,     // (T*F) x H  (transposed W1)
    const float* __restrict__ b1,    // T*F
    const float* __restrict__ norm_w,// F
    f16* __restrict__ xn)            // (B*T) x F
{
  const int b0   = blockIdx.x * 16;
  const int t    = blockIdx.y;
  const int tid  = threadIdx.x;
  const int w    = tid >> 5;
  const int lane = tid & 31;
  const int half = lane >> 4;
  const int l16  = lane & 15;
  const int fbase = w * 64;

  v8f acc[4];
#pragma unroll
  for (int tl = 0; tl < 4; ++tl) acc[tl] = (v8f)0.0f;

  const f16* arow = feath + (b0 + l16) * H_DIM;
  const f16* bcol[4];
#pragma unroll
  for (int tl = 0; tl < 4; ++tl)
    bcol[tl] = w1t + (size_t)(t * F_DIM + fbase + tl * 16 + l16) * H_DIM;

  for (int k0 = 0; k0 < H_DIM; k0 += 32) {
    const v16h a = load_a_frag(arow + k0, half);
#pragma unroll
    for (int tl = 0; tl < 4; ++tl) {
      const v16h b = load_b_frag(bcol[tl] + k0, half);
      acc[tl] = wmma_f16(a, b, acc[tl]);
    }
  }

  // Epilogue: + b1, per-row sum of squares (row m = half*8 + r).
  float ss[8];
#pragma unroll
  for (int r = 0; r < 8; ++r) ss[r] = 0.0f;
#pragma unroll
  for (int tl = 0; tl < 4; ++tl) {
    const float b1v = b1[t * F_DIM + fbase + tl * 16 + l16];
#pragma unroll
    for (int r = 0; r < 8; ++r) {
      const float v = acc[tl][r] + b1v;
      acc[tl][r] = v;
      ss[r] += v * v;
    }
  }
  // Reduce across the 16 lanes of each half (masks 1,2,4,8 stay inside a half).
#pragma unroll
  for (int m = 1; m <= 8; m <<= 1)
#pragma unroll
    for (int r = 0; r < 8; ++r) ss[r] += __shfl_xor(ss[r], m, 32);

  __shared__ float smrow[8 * 16];
  if (l16 == 0) {
#pragma unroll
    for (int r = 0; r < 8; ++r) smrow[w * 16 + half * 8 + r] = ss[r];
  }
  __syncthreads();

  float inv[8];
#pragma unroll
  for (int r = 0; r < 8; ++r) {
    float tot = 0.0f;
#pragma unroll
    for (int w2 = 0; w2 < 8; ++w2) tot += smrow[w2 * 16 + half * 8 + r];
    inv[r] = rsqrtf(tot * (1.0f / F_DIM) + RMS_EPS_C);
  }

#pragma unroll
  for (int tl = 0; tl < 4; ++tl) {
    const int f = fbase + tl * 16 + l16;
    const float nw = norm_w[f];
#pragma unroll
    for (int r = 0; r < 8; ++r) {
      const int m = half * 8 + r;
      const float v = acc[tl][r] * inv[r] * nw;
      xn[((size_t)(b0 + m) * T_DIM + t) * F_DIM + f] = (f16)v;
    }
  }
}

// ---------------------------------------------------------------------------
// Kernel B: logits = xn @ W2 + b2, fused softmax over T, CDF, censoring and
// log-loss. M-register-blocked: each WG covers 64 rows (4 b-pairs) x 256 cols,
// so every B fragment is reused by 4 WMMAs (4x less L2 traffic on W2t).
// Lane slot r of row-tile mt holds logit(b = 8*by + 2*mt + half, t = r, k).
// Grid: (K/256, B/8). 256 threads = 8 waves; wave w owns 2 col-tiles.
// ---------------------------------------------------------------------------
#define MT 4   // row-tiles (of 16) per workgroup
__global__ __launch_bounds__(256) void gemm2_loss(
    const f16* __restrict__ xn,       // (B*T) x F
    const f16* __restrict__ w2t,      // K x F (transposed W2)
    const float* __restrict__ b2,     // K
    const unsigned char* __restrict__ is_event,  // B x T x K (bool)
    const unsigned char* __restrict__ is_cens,   // B x K (bool)
    const float* __restrict__ ctr,    // B x T x K
    float* __restrict__ pLoss, float* __restrict__ pCnt)
{
  const int n0   = blockIdx.x * 256;
  const int by   = blockIdx.y;
  const int r0   = by * (16 * MT);   // 64 consecutive xn rows = batches 8by..8by+7
  const int tid  = threadIdx.x;
  const int w    = tid >> 5;
  const int lane = tid & 31;
  const int half = lane >> 4;
  const int l16  = lane & 15;

  v8f acc[MT][2];
#pragma unroll
  for (int mt = 0; mt < MT; ++mt) { acc[mt][0] = (v8f)0.0f; acc[mt][1] = (v8f)0.0f; }

  const f16* arow[MT];
#pragma unroll
  for (int mt = 0; mt < MT; ++mt)
    arow[mt] = xn + (size_t)(r0 + mt * 16 + l16) * F_DIM;

  int kcol[2];
  const f16* bcol[2];
#pragma unroll
  for (int j = 0; j < 2; ++j) {
    kcol[j] = n0 + w * 32 + j * 16 + l16;
    bcol[j] = w2t + (size_t)kcol[j] * F_DIM;
  }

  for (int f0 = 0; f0 < F_DIM; f0 += 32) {
    v16h a[MT];
#pragma unroll
    for (int mt = 0; mt < MT; ++mt) a[mt] = load_a_frag(arow[mt] + f0, half);
#pragma unroll
    for (int j = 0; j < 2; ++j) {
      const v16h b = load_b_frag(bcol[j] + f0, half);
#pragma unroll
      for (int mt = 0; mt < MT; ++mt) acc[mt][j] = wmma_f16(a[mt], b, acc[mt][j]);
    }
  }

  float lsum = 0.0f, lcnt = 0.0f;

#pragma unroll
  for (int j = 0; j < 2; ++j) {
    const int k = kcol[j];
    const float bias = b2[k];
#pragma unroll
    for (int mt = 0; mt < MT; ++mt) {
      const int b = by * (2 * MT) + mt * 2 + half;
      float lg[8], mx = -1e30f;
#pragma unroll
      for (int r = 0; r < 8; ++r) { lg[r] = acc[mt][j][r] + bias; mx = fmaxf(mx, lg[r]); }
      float e[8], s = 0.0f;
#pragma unroll
      for (int r = 0; r < 8; ++r) { e[r] = expf(lg[r] - mx); s += e[r]; }
      const float invs = 1.0f / s;

      const bool cen = is_cens[(size_t)b * K_DIM + k] != 0;
      const size_t base = ((size_t)b * T_DIM) * K_DIM + k;

      float integ = 1.0f, cum = 0.0f;
#pragma unroll
      for (int tt = 0; tt < 8; ++tt) {
        const float pt = e[tt] * invs;
        const float ps = fminf(fmaxf(pt, EPS_C), ONE_M_EPS);
        float sel;
        if (cen) {
          const float is_ = fminf(fmaxf(integ, EPS_C), ONE_M_EPS);
          const float c   = ctr[base + (size_t)tt * K_DIM];
          sel = fminf(fmaxf(is_ - c * ps, EPS_C), ONE_M_EPS);
        } else {
          sel = ps;
        }
        if (is_event[base + (size_t)tt * K_DIM]) { lsum += logf(sel); lcnt += 1.0f; }
        cum += pt;
        integ = 1.0f - cum;
      }
    }
  }

  // Deterministic block reduction -> per-WG partials.
#pragma unroll
  for (int m = 1; m <= 16; m <<= 1) {
    lsum += __shfl_xor(lsum, m, 32);
    lcnt += __shfl_xor(lcnt, m, 32);
  }
  __shared__ float sL[8], sC[8];
  if (lane == 0) { sL[w] = lsum; sC[w] = lcnt; }
  __syncthreads();
  if (tid == 0) {
    float S = 0.0f, C = 0.0f;
#pragma unroll
    for (int i = 0; i < 8; ++i) { S += sL[i]; C += sC[i]; }
    const int wg = by * gridDim.x + blockIdx.x;
    pLoss[wg] = S;
    pCnt[wg]  = C;
  }
}

// ---------------------------------------------------------------------------
// Kernel C: deterministic final reduction of the partials.
// ---------------------------------------------------------------------------
__global__ __launch_bounds__(256) void finalize(
    const float* __restrict__ pL, const float* __restrict__ pC,
    float* __restrict__ out, int n)
{
  __shared__ float sL[256], sC[256];
  const int tid = threadIdx.x;
  float S = 0.0f, C = 0.0f;
  for (int i = tid; i < n; i += 256) { S += pL[i]; C += pC[i]; }
  sL[tid] = S; sC[tid] = C;
  __syncthreads();
  for (int s = 128; s > 0; s >>= 1) {
    if (tid < s) { sL[tid] += sL[tid + s]; sC[tid] += sC[tid + s]; }
    __syncthreads();
  }
  if (tid == 0) out[0] = -sL[0] / (sC[0] * LN_T);
}

// ---------------------------------------------------------------------------
extern "C" void kernel_launch(void* const* d_in, const int* in_sizes, int n_in,
                              void* d_out, int out_size, void* d_ws, size_t ws_size,
                              hipStream_t stream) {
  (void)in_sizes; (void)n_in; (void)out_size; (void)ws_size;

  const float* features = (const float*)d_in[0];   // B x H
  const float* W1       = (const float*)d_in[1];   // H x (T*F)
  const float* b1       = (const float*)d_in[2];   // T*F
  const float* norm_w   = (const float*)d_in[3];   // F
  const float* W2       = (const float*)d_in[4];   // F x K
  const float* b2       = (const float*)d_in[5];   // K
  const unsigned char* is_event = (const unsigned char*)d_in[6];  // B x T x K
  const unsigned char* is_cens  = (const unsigned char*)d_in[7];  // B x K
  const float* ctr      = (const float*)d_in[8];   // B x T x K
  float* out = (float*)d_out;

  // Workspace layout (f16 staging + partials), 256B-aligned regions.
  char* ws = (char*)d_ws;
  size_t off = 0;
  f16* feath = (f16*)(ws + off); off += (size_t)B_DIM * H_DIM * 2;          // 0.75 MB
  f16* w1t   = (f16*)(ws + off); off += (size_t)(T_DIM * F_DIM) * H_DIM * 2;// 6 MB
  f16* w2t   = (f16*)(ws + off); off += (size_t)K_DIM * F_DIM * 2;          // 8 MB
  f16* xnb   = (f16*)(ws + off); off += (size_t)(B_DIM * T_DIM) * F_DIM * 2;// 4 MB
  off = (off + 255) & ~(size_t)255;
  const int NPART = (K_DIM / 256) * (B_DIM / (2 * MT));   // 32 * 64 = 2048
  float* pL  = (float*)(ws + off); off += (size_t)NPART * 4;
  float* pC  = (float*)(ws + off); off += (size_t)NPART * 4;

  // 1) Convert/transpose operands to f16 in WMMA-friendly layouts.
  cvt_f16   <<<1024, 256, 0, stream>>>(features, feath, B_DIM * H_DIM);
  cvt_f16_t <<<4096, 256, 0, stream>>>(W1, w1t, H_DIM, T_DIM * F_DIM);
  cvt_f16_t <<<4096, 256, 0, stream>>>(W2, w2t, F_DIM, K_DIM);

  // 2) GEMM1 + RMS norm -> xn (f16).
  gemm1_rms <<<dim3(B_DIM / 16, T_DIM), 256, 0, stream>>>(feath, w1t, b1,
                                                          norm_w, xnb);

  // 3) GEMM2 (M-blocked) fused with softmax-over-T + censored log-loss.
  gemm2_loss <<<dim3(K_DIM / 256, B_DIM / (2 * MT)), 256, 0, stream>>>(
      xnb, w2t, b2, is_event, is_cens, ctr, pL, pC);

  // 4) Deterministic scalar finalize.
  finalize <<<1, 256, 0, stream>>>(pL, pC, out, NPART);
}
